// Spatial_Tempoal_Aware_71863392796931
// MI455X (gfx1250) — matrix operand order repaired and checked by reading, not compiled, not run on hardware
//
#include <hip/hip_runtime.h>
#include <hip/hip_bf16.h>
#include <math.h>

typedef _Float16 half_t;
typedef __attribute__((ext_vector_type(16))) _Float16 v16h;
typedef __attribute__((ext_vector_type(8)))  float    v8f;

#define B_  16
#define L_  12
#define N_  320
#define C_  64
#define H_  8
#define FF_ 256
#define CL_ 20
#define S_  60
#define T_  61440           // B*L*N
#define BL_ 192             // B*L

enum { MODE_F16 = 0, MODE_F16_RELU = 1, MODE_F32 = 2 };

// ---------------------------------------------------------------------------
// WMMA fragment helpers (CDNA5 16-bit A 16x32 / B 32x16 / f32 C 16x16 layouts)
// ---------------------------------------------------------------------------
__device__ __forceinline__ int a_frag_k(int i, int hi) {
    // lanes 0-15: VGPR0-7 -> K=0..15
    // lanes16-31: VGPR0-7 -> K=16..31
    return hi * 16 + 2 * i;
}

// ---------------------------------------------------------------------------
// Temporal_Aware conv branches: ReplPad(3,3) -> Conv(1,5) -> LeakyReLU ->
// Conv(1,3) -> tanh, two branches.  One block per (b,n), thread = out channel.
// ---------------------------------------------------------------------------
__global__ __launch_bounds__(64)
void conv_branch_kernel(const float* __restrict__ x,
                        const float* __restrict__ w1a, const float* __restrict__ b1a,
                        const float* __restrict__ w1b, const float* __restrict__ b1b,
                        const float* __restrict__ w2a, const float* __restrict__ b2a,
                        const float* __restrict__ w2b, const float* __restrict__ b2b,
                        float* __restrict__ x1, float* __restrict__ x2,
                        half_t* __restrict__ xo)
{
    __shared__ float xt[64 * 12];
    __shared__ float y1[64 * 14];
    const int bn = blockIdx.x;            // b*N + n
    const int b  = bn / N_;
    const int n  = bn % N_;
    const int co = threadIdx.x;           // 0..63

    for (int l = 0; l < L_; ++l)
        xt[co * 12 + l] = x[((size_t)(b * L_ + l) * N_ + n) * C_ + co];
    __syncthreads();

    for (int br = 0; br < 2; ++br) {
        const float* wa = br ? w2a : w1a;
        const float* ba = br ? b2a : b1a;
        const float* wb = br ? w2b : w1b;
        const float* bb = br ? b2b : b1b;

        // conv (1,5) over replication-padded length 18 -> 14, LeakyReLU(0.01)
        for (int lp = 0; lp < 14; ++lp) {
            float s = ba[co];
            for (int ci = 0; ci < 64; ++ci) {
                const float* wr = wa + (co * 64 + ci) * 5;
                const float* xr = xt + ci * 12;
#pragma unroll
                for (int k = 0; k < 5; ++k) {
                    int xi = lp + k - 3;
                    xi = xi < 0 ? 0 : (xi > 11 ? 11 : xi);
                    s += wr[k] * xr[xi];
                }
            }
            y1[co * 14 + lp] = (s > 0.f) ? s : 0.01f * s;
        }
        __syncthreads();

        // conv (1,3) valid: 14 -> 12, tanh
        for (int l = 0; l < L_; ++l) {
            float s = bb[co];
            for (int ci = 0; ci < 64; ++ci) {
                const float* wr = wb + (co * 64 + ci) * 3;
                const float* yr = y1 + ci * 14 + l;
                s += wr[0] * yr[0] + wr[1] * yr[1] + wr[2] * yr[2];
            }
            float v = tanhf(s);
            size_t t = (size_t)(b * L_ + l) * N_ + n;
            (br ? x2 : x1)[t * C_ + co] = v;
            xo[t * 128 + br * 64 + co]  = (half_t)v;   // concat([x1,x2],-1)
        }
        __syncthreads();
    }
}

// ---------------------------------------------------------------------------
// Generic WMMA GEMM:  out[M,N] = mode( A_f16[M,K] @ W_f32[K,N] + bias )
// block = 128 threads (4 waves), macro tile 64 x 16, W tile staged in LDS
// transposed as f16 so B-fragment pairs are contiguous.
// ---------------------------------------------------------------------------
__global__ __launch_bounds__(128)
void gemm_wmma_kernel(const half_t* __restrict__ A, const float* __restrict__ W,
                      const float* __restrict__ bias, void* __restrict__ out,
                      int K, int N, int mode)
{
    __shared__ half_t lw[16 * 256];     // K <= 256
    const int tid  = threadIdx.x;
    const int lane = tid & 31;
    const int wave = tid >> 5;
    const int n0   = blockIdx.y * 16;

    for (int idx = tid; idx < K * 16; idx += 128) {
        int k = idx >> 4, n = idx & 15;
        lw[n * K + k] = (half_t)W[k * N + n0 + n];
    }
    __syncthreads();

    const int m0   = blockIdx.x * 64 + wave * 16;
    const int nlo  = lane & 15;
    const int hi   = lane >> 4;

    float bval = bias[n0 + nlo];
    v8f acc;
#pragma unroll
    for (int r = 0; r < 8; ++r) acc[r] = bval;

    const half_t* arow = A + (size_t)(m0 + nlo) * K;   // row m = m0 + (lane&15)
    const half_t* brow = lw + nlo * K;                 // col n = lane&15

    for (int k0 = 0; k0 < K; k0 += 32) {
        v16h a, b;
#pragma unroll
        for (int i = 0; i < 8; ++i) {
            int ka = k0 + a_frag_k(i, hi);
            a[2 * i]     = arow[ka];
            a[2 * i + 1] = arow[ka + 1];
            int kb = k0 + hi * 16 + 2 * i;
            b[2 * i]     = brow[kb];
            b[2 * i + 1] = brow[kb + 1];
        }
        acc = __builtin_amdgcn_wmma_f32_16x16x32_f16(false, a, false, b,
                                                     (short)0, acc, false, false);
    }

#pragma unroll
    for (int r = 0; r < 8; ++r) {
        int m = m0 + r + 8 * hi;
        int n = n0 + nlo;
        float v = acc[r];
        if (mode == MODE_F16)
            ((half_t*)out)[(size_t)m * N + n] = (half_t)v;
        else if (mode == MODE_F16_RELU)
            ((half_t*)out)[(size_t)m * N + n] = (half_t)(v > 0.f ? v : 0.f);
        else
            ((float*)out)[(size_t)m * N + n] = v;
    }
}

// ---------------------------------------------------------------------------
// Temporal attention over the node axis, per (b,l,h): q,k,v (320 x 16).
// grid.x = BL*H, grid.y = 10 (two 16-row q chunks per block, one per wave).
// ---------------------------------------------------------------------------
__global__ __launch_bounds__(64)
void temporal_attn_kernel(const half_t* __restrict__ qt, const half_t* __restrict__ kt,
                          const half_t* __restrict__ vt, half_t* __restrict__ ot)
{
    __shared__ half_t Kl[N_ * 16];
    __shared__ half_t Vl[N_ * 16];
    __shared__ float  Sb[2][16 * N_];

    const int tid  = threadIdx.x;
    const int lane = tid & 31;
    const int wave = tid >> 5;
    const int blh  = blockIdx.x;
    const int h    = blh % H_;
    const int bl   = blh / H_;
    const int t0   = bl * N_;

    for (int idx = tid; idx < N_ * 16; idx += 64) {
        int node = idx >> 4, d = idx & 15;
        Kl[idx] = kt[(size_t)(t0 + node) * 128 + h * 16 + d];
        Vl[idx] = vt[(size_t)(t0 + node) * 128 + h * 16 + d];
    }
    __syncthreads();

    const int chunk = blockIdx.y * 2 + wave;
    const int m0    = chunk * 16;
    const int nlo   = lane & 15;
    const int hi    = lane >> 4;

    // A fragment: 16 q rows, K = 16 zero-padded to 32
    const half_t* qrow = qt + (size_t)(t0 + m0 + nlo) * 128 + h * 16;
    v16h aq;
#pragma unroll
    for (int i = 0; i < 8; ++i) {
        int k = a_frag_k(i, hi);
        aq[2 * i]     = (k < 16) ? qrow[k]     : (half_t)0.f;
        aq[2 * i + 1] = (k < 16) ? qrow[k + 1] : (half_t)0.f;
    }

    float* srow_base = Sb[wave];
    for (int j = 0; j < N_ / 16; ++j) {              // 20 key tiles
        v16h b;
#pragma unroll
        for (int i = 0; i < 8; ++i) {
            int k = hi * 16 + 2 * i;
            if (k < 16) {
                b[2 * i]     = Kl[(j * 16 + nlo) * 16 + k];
                b[2 * i + 1] = Kl[(j * 16 + nlo) * 16 + k + 1];
            } else {
                b[2 * i] = (half_t)0.f; b[2 * i + 1] = (half_t)0.f;
            }
        }
        v8f acc = {0.f, 0.f, 0.f, 0.f, 0.f, 0.f, 0.f, 0.f};
        acc = __builtin_amdgcn_wmma_f32_16x16x32_f16(false, aq, false, b,
                                                     (short)0, acc, false, false);
#pragma unroll
        for (int r = 0; r < 8; ++r)
            srow_base[(r + 8 * hi) * N_ + j * 16 + nlo] = acc[r] * 0.25f;  // 1/sqrt(16)
    }
    __syncthreads();

    // row softmax over 320 keys: lanes 0..15 of each wave take one row each
    if (lane < 16) {
        float* row = srow_base + lane * N_;
        float mx = -1e30f;
        for (int i = 0; i < N_; ++i) mx = fmaxf(mx, row[i]);
        float sum = 0.f;
        for (int i = 0; i < N_; ++i) { float e = __expf(row[i] - mx); row[i] = e; sum += e; }
        float inv = 1.f / sum;
        for (int i = 0; i < N_; ++i) row[i] *= inv;
    }
    __syncthreads();

    // O = softmax(S) @ V   (16 x 320 @ 320 x 16)
    v8f oacc = {0.f, 0.f, 0.f, 0.f, 0.f, 0.f, 0.f, 0.f};
    const float* srow = srow_base + nlo * N_;
    for (int k0 = 0; k0 < N_; k0 += 32) {
        v16h a, b;
#pragma unroll
        for (int i = 0; i < 8; ++i) {
            int ka = a_frag_k(i, hi);
            a[2 * i]     = (half_t)srow[k0 + ka];
            a[2 * i + 1] = (half_t)srow[k0 + ka + 1];
            int kb = hi * 16 + 2 * i;
            b[2 * i]     = Vl[(k0 + kb) * 16 + nlo];
            b[2 * i + 1] = Vl[(k0 + kb + 1) * 16 + nlo];
        }
        oacc = __builtin_amdgcn_wmma_f32_16x16x32_f16(false, a, false, b,
                                                      (short)0, oacc, false, false);
    }
#pragma unroll
    for (int r = 0; r < 8; ++r)
        ot[(size_t)(t0 + m0 + r + 8 * hi) * 128 + h * 16 + nlo] = (half_t)oacc[r];
}

// ---------------------------------------------------------------------------
// z = sigmoid(gate);  t_out = tanh(x1)*z + tanh(x2)*(1-z)   (f16 out)
// ---------------------------------------------------------------------------
__global__ __launch_bounds__(256)
void gate_kernel(const float* __restrict__ gin, const float* __restrict__ x1,
                 const float* __restrict__ x2, half_t* __restrict__ tout, int total)
{
    int i = blockIdx.x * 256 + threadIdx.x;
    if (i >= total) return;
    float z = 1.f / (1.f + __expf(-gin[i]));
    float t = tanhf(x1[i]) * z + tanhf(x2[i]) * (1.f - z);
    tout[i] = (half_t)t;
}

// ---------------------------------------------------------------------------
// LayerNorm over C=64: out = LN(a + b) * g + beta; one token per thread
// ---------------------------------------------------------------------------
__global__ __launch_bounds__(256)
void ln_kernel(const float* __restrict__ a, const float* __restrict__ b,
               const float* __restrict__ g, const float* __restrict__ beta,
               float* __restrict__ out32, half_t* __restrict__ out16, int Ttok)
{
    int t = blockIdx.x * 256 + threadIdx.x;
    if (t >= Ttok) return;
    size_t base = (size_t)t * C_;
    float v[C_];
    float mean = 0.f;
#pragma unroll
    for (int i = 0; i < C_; ++i) { v[i] = a[base + i] + b[base + i]; mean += v[i]; }
    mean *= (1.f / C_);
    float var = 0.f;
#pragma unroll
    for (int i = 0; i < C_; ++i) { float d = v[i] - mean; var += d * d; }
    var *= (1.f / C_);
    float rstd = rsqrtf(var + 1e-5f);
#pragma unroll
    for (int i = 0; i < C_; ++i) {
        float o = (v[i] - mean) * rstd * g[i] + beta[i];
        out32[base + i] = o;
        if (out16) out16[base + i] = (half_t)o;
    }
}

// ---------------------------------------------------------------------------
// AdaptiveAvgPool over 16 nodes per cluster: kc/vc[b,l,cl,c] (f16 out)
// ---------------------------------------------------------------------------
__global__ __launch_bounds__(256)
void pool_kernel(const float* __restrict__ kf, const float* __restrict__ vf,
                 half_t* __restrict__ kc, half_t* __restrict__ vc, int total)
{
    int idx = blockIdx.x * 256 + threadIdx.x;
    if (idx >= total) return;
    int c  = idx & 63;
    int cl = (idx >> 6) % CL_;
    int bl = idx / (64 * CL_);
    size_t base = ((size_t)bl * N_ + cl * 16) * C_ + c;
    float sk = 0.f, sv = 0.f;
    for (int j = 0; j < 16; ++j) { sk += kf[base + j * C_]; sv += vf[base + j * C_]; }
    kc[idx] = (half_t)(sk * (1.f / 16.f));
    vc[idx] = (half_t)(sv * (1.f / 16.f));
}

// ---------------------------------------------------------------------------
// External node-memory attention: softmax over the node axis (n = 320).
// grid.x = BL*H, block = 320 threads (thread = node n).
// exa = (softmax_n(ev @ U1) @ U2) * 0.5 + ev
// ---------------------------------------------------------------------------
__global__ __launch_bounds__(320)
void extmem_kernel(const float* __restrict__ ev, const float* __restrict__ U1,
                   const float* __restrict__ U2, float* __restrict__ exa)
{
    __shared__ float red[N_];
    __shared__ float U1l[8 * S_];
    __shared__ float U2l[S_ * 8];
    const int tid = threadIdx.x;
    const int blh = blockIdx.x;
    const int h   = blh % H_;
    const int bl  = blh / H_;
    const size_t t0 = (size_t)bl * N_;

    for (int i = tid; i < 8 * S_; i += N_) { U1l[i] = U1[i]; U2l[i] = U2[i]; }

    float myev[8];
#pragma unroll
    for (int d = 0; d < 8; ++d) myev[d] = ev[(t0 + tid) * C_ + h * 8 + d];
    __syncthreads();

    float ex[8] = {0.f, 0.f, 0.f, 0.f, 0.f, 0.f, 0.f, 0.f};
    for (int s = 0; s < S_; ++s) {
        float lg = 0.f;
#pragma unroll
        for (int d = 0; d < 8; ++d) lg += myev[d] * U1l[d * S_ + s];

        red[tid] = lg; __syncthreads();
        for (int off = 256; off > 0; off >>= 1) {
            if (tid < off && tid + off < N_) red[tid] = fmaxf(red[tid], red[tid + off]);
            __syncthreads();
        }
        float mx = red[0]; __syncthreads();

        float e = __expf(lg - mx);
        red[tid] = e; __syncthreads();
        for (int off = 256; off > 0; off >>= 1) {
            if (tid < off && tid + off < N_) red[tid] += red[tid + off];
            __syncthreads();
        }
        float ea = e / red[0]; __syncthreads();

#pragma unroll
        for (int d = 0; d < 8; ++d) ex[d] += ea * U2l[s * 8 + d];
    }
#pragma unroll
    for (int d = 0; d < 8; ++d)
        exa[(t0 + tid) * C_ + h * 8 + d] = ex[d] * 0.5f + myev[d];
}

// ---------------------------------------------------------------------------
// Spa attention: per (b,l), thread = node; 20 clustered keys/values + adp_pos.
// spa = softmax(q.kc/sqrt(8) + adp_pos) @ vc + exa   (f16 out)
// ---------------------------------------------------------------------------
__global__ __launch_bounds__(320)
void spa_attn_kernel(const half_t* __restrict__ qs, const half_t* __restrict__ kc,
                     const half_t* __restrict__ vc, const float* __restrict__ adp,
                     const float* __restrict__ exa, half_t* __restrict__ spa)
{
    __shared__ half_t kcl[CL_ * C_];
    __shared__ half_t vcl[CL_ * C_];
    const int tid = threadIdx.x;
    const int bl  = blockIdx.x;
    for (int i = tid; i < CL_ * C_; i += N_) {
        kcl[i] = kc[(size_t)bl * CL_ * C_ + i];
        vcl[i] = vc[(size_t)bl * CL_ * C_ + i];
    }
    __syncthreads();

    const size_t t = (size_t)bl * N_ + tid;
    for (int h = 0; h < H_; ++h) {
        float q[8];
#pragma unroll
        for (int d = 0; d < 8; ++d) q[d] = (float)qs[t * C_ + h * 8 + d];
        float sc[CL_], mx = -1e30f;
        for (int m = 0; m < CL_; ++m) {
            float s = 0.f;
#pragma unroll
            for (int d = 0; d < 8; ++d) s += q[d] * (float)kcl[m * C_ + h * 8 + d];
            s = s * 0.35355339059f + adp[tid * CL_ + m];   // 1/sqrt(8)
            sc[m] = s; mx = fmaxf(mx, s);
        }
        float sum = 0.f;
        for (int m = 0; m < CL_; ++m) { sc[m] = __expf(sc[m] - mx); sum += sc[m]; }
        float inv = 1.f / sum;
#pragma unroll
        for (int d = 0; d < 8; ++d) {
            float o = 0.f;
            for (int m = 0; m < CL_; ++m) o += sc[m] * (float)vcl[m * C_ + h * 8 + d];
            o = o * inv + exa[t * C_ + h * 8 + d];
            spa[t * C_ + h * 8 + d] = (half_t)o;
        }
    }
}

// ---------------------------------------------------------------------------
extern "C" void kernel_launch(void* const* d_in, const int* in_sizes, int n_in,
                              void* d_out, int out_size, void* d_ws, size_t ws_size,
                              hipStream_t stream)
{
    (void)in_sizes; (void)n_in; (void)out_size; (void)ws_size;
    // inputs in setup_inputs() dict order: x, spatial_embedding, params...
    const float* x      = (const float*)d_in[0];
    // d_in[1] = spatial_embedding (unused by the reference forward)
    const float* c1a_w  = (const float*)d_in[2];
    const float* c1a_b  = (const float*)d_in[3];
    const float* c1b_w  = (const float*)d_in[4];
    const float* c1b_b  = (const float*)d_in[5];
    const float* c2a_w  = (const float*)d_in[6];
    const float* c2a_b  = (const float*)d_in[7];
    const float* c2b_w  = (const float*)d_in[8];
    const float* c2b_b  = (const float*)d_in[9];
    const float* tq_w   = (const float*)d_in[10];
    const float* tq_b   = (const float*)d_in[11];
    const float* tk_w   = (const float*)d_in[12];
    const float* tk_b   = (const float*)d_in[13];
    const float* tv_w   = (const float*)d_in[14];
    const float* tv_b   = (const float*)d_in[15];
    const float* to_w   = (const float*)d_in[16];
    const float* to_b   = (const float*)d_in[17];
    const float* sq_w   = (const float*)d_in[18];
    const float* sq_b   = (const float*)d_in[19];
    const float* sk_w   = (const float*)d_in[20];
    const float* sk_b   = (const float*)d_in[21];
    const float* sv_w   = (const float*)d_in[22];
    const float* sv_b   = (const float*)d_in[23];
    const float* se_w   = (const float*)d_in[24];
    const float* se_b   = (const float*)d_in[25];
    const float* so_w   = (const float*)d_in[26];
    const float* so_b   = (const float*)d_in[27];
    const float* adp    = (const float*)d_in[28];
    const float* U1     = (const float*)d_in[29];
    const float* U2     = (const float*)d_in[30];
    const float* f1w1   = (const float*)d_in[31];
    const float* f1b1   = (const float*)d_in[32];
    const float* f1w2   = (const float*)d_in[33];
    const float* f1b2   = (const float*)d_in[34];
    const float* f2w1   = (const float*)d_in[35];
    const float* f2b1   = (const float*)d_in[36];
    const float* f2w2   = (const float*)d_in[37];
    const float* f2b2   = (const float*)d_in[38];
    const float* ln2_g  = (const float*)d_in[39];
    const float* ln2_b  = (const float*)d_in[40];

    // ---- workspace carving (TCb = T*64*4 bytes); overlays honor liveness ----
    char* w = (char*)d_ws;
    const size_t TCb = (size_t)T_ * C_ * 4;           // 15.7 MB
    float*  x1      = (float*) (w + 0 * TCb);
    float*  x2      = (float*) (w + 1 * TCb);
    half_t* xo_h    = (half_t*)(w + 2 * TCb);         // [T,128] f16
    half_t* qt      = (half_t*)(w + 3 * TCb);
    half_t* kt      = (half_t*)(w + 4 * TCb);
    half_t* vt      = (half_t*)(w + 5 * TCb);
    half_t* ot      = (half_t*)(w + 6 * TCb);
    float*  gate_in = (float*) (w + 7 * TCb);
    half_t* tout_h  = (half_t*)(w + 8 * TCb);         // TCb/2 used
    float*  resid   = (float*) (w + (size_t)(8.5 * TCb));
    half_t* resid_h = (half_t*)(w + (size_t)(9.5 * TCb));
    half_t* qs_h    = (half_t*)(w + 10 * TCb);
    half_t* kc_h    = (half_t*)(w + (size_t)(10.5 * TCb));
    half_t* vc_h    = kc_h + (size_t)BL_ * CL_ * C_;
    // overlays (dead regions reused):
    half_t* h_big   = (half_t*)(w + 3 * TCb);         // [T,256] f16 over qt+kt
    float*  ffo     = (float*) (w + 5 * TCb);         // [T,64] f32 over vt
    float*  kf      = (float*) (w + 6 * TCb);         // over ot
    float*  vf      = (float*) (w + 7 * TCb);         // over gate_in
    float*  evb     = (float*) (w + 0 * TCb);         // over x1
    float*  exa     = (float*) (w + 1 * TCb);         // over x2
    half_t* spa_h   = (half_t*)(w + 2 * TCb);         // over xo (lo half)
    half_t* so_h    = (half_t*)(w + (size_t)(2.5 * TCb)); // over xo (hi half)

    float*  outp    = (float*)d_out;

    const int elems = T_ * C_;

    // 1) Temporal_Aware conv branches -> x1, x2 (f32) and xo = concat (f16)
    conv_branch_kernel<<<dim3(B_ * N_), dim3(64), 0, stream>>>(
        x, c1a_w, c1a_b, c1b_w, c1b_b, c2a_w, c2a_b, c2b_w, c2b_b, x1, x2, xo_h);

    // 2) temporal q/k/v projections (K=128, N=128)
    gemm_wmma_kernel<<<dim3(T_ / 64, 8), 128, 0, stream>>>(xo_h, tq_w, tq_b, qt, 128, 128, MODE_F16);
    gemm_wmma_kernel<<<dim3(T_ / 64, 8), 128, 0, stream>>>(xo_h, tk_w, tk_b, kt, 128, 128, MODE_F16);
    gemm_wmma_kernel<<<dim3(T_ / 64, 8), 128, 0, stream>>>(xo_h, tv_w, tv_b, vt, 128, 128, MODE_F16);

    // 3) temporal attention over nodes -> ot (f16 [T,128])
    temporal_attn_kernel<<<dim3(BL_ * H_, 10), 64, 0, stream>>>(qt, kt, vt, ot);

    // 4) out_proj 2C->C (f32), then 5) sigmoid gate -> t_out (f16)
    gemm_wmma_kernel<<<dim3(T_ / 64, 4), 128, 0, stream>>>(ot, to_w, to_b, gate_in, 128, 64, MODE_F32);
    gate_kernel<<<dim3((elems + 255) / 256), 256, 0, stream>>>(gate_in, x1, x2, tout_h, elems);

    // 6,7) FFN1 + 8) residual LayerNorm
    gemm_wmma_kernel<<<dim3(T_ / 64, 16), 128, 0, stream>>>(tout_h, f1w1, f1b1, h_big, 64, 256, MODE_F16_RELU);
    gemm_wmma_kernel<<<dim3(T_ / 64, 4), 128, 0, stream>>>(h_big, f1w2, f1b2, ffo, 256, 64, MODE_F32);
    ln_kernel<<<dim3((T_ + 255) / 256), 256, 0, stream>>>(x, ffo, ln2_g, ln2_b, resid, resid_h, T_);

    // 9-12) spa projections
    gemm_wmma_kernel<<<dim3(T_ / 64, 4), 128, 0, stream>>>(resid_h, sq_w, sq_b, qs_h, 64, 64, MODE_F16);
    gemm_wmma_kernel<<<dim3(T_ / 64, 4), 128, 0, stream>>>(resid_h, sk_w, sk_b, kf,   64, 64, MODE_F32);
    gemm_wmma_kernel<<<dim3(T_ / 64, 4), 128, 0, stream>>>(resid_h, sv_w, sv_b, vf,   64, 64, MODE_F32);
    gemm_wmma_kernel<<<dim3(T_ / 64, 4), 128, 0, stream>>>(resid_h, se_w, se_b, evb,  64, 64, MODE_F32);

    // 13) cluster pooling
    pool_kernel<<<dim3((BL_ * CL_ * C_ + 255) / 256), 256, 0, stream>>>(
        kf, vf, kc_h, vc_h, BL_ * CL_ * C_);

    // 14) external node-memory attention (softmax over n)
    extmem_kernel<<<dim3(BL_ * H_), 320, 0, stream>>>(evb, U1, U2, exa);

    // 15) clustered spa attention + exa
    spa_attn_kernel<<<dim3(BL_), 320, 0, stream>>>(qs_h, kc_h, vc_h, adp, exa, spa_h);

    // 16) spa out_proj, 17,18) FFN2, 19) final residual LayerNorm -> d_out
    gemm_wmma_kernel<<<dim3(T_ / 64, 4), 128, 0, stream>>>(spa_h, so_w, so_b, so_h, 64, 64, MODE_F16);
    gemm_wmma_kernel<<<dim3(T_ / 64, 16), 128, 0, stream>>>(so_h, f2w1, f2b1, h_big, 64, 256, MODE_F16_RELU);
    gemm_wmma_kernel<<<dim3(T_ / 64, 4), 128, 0, stream>>>(h_big, f2w2, f2b2, ffo, 256, 64, MODE_F32);
    ln_kernel<<<dim3((T_ + 255) / 256), 256, 0, stream>>>(resid, ffo, ln2_g, ln2_b, outp, (half_t*)nullptr, T_);
}
